// SparseMHA_56135222558972
// MI455X (gfx1250) — matrix operand order repaired and checked by reading, not compiled, not run on hardware
//
#include <hip/hip_runtime.h>
#include <hip/hip_bf16.h>

#define HID 256
#define HEADS 8
#define HD 32

typedef __attribute__((ext_vector_type(16))) _Float16 v16h;
typedef __attribute__((ext_vector_type(8)))  _Float16 v8h;
typedef __attribute__((ext_vector_type(8)))  float    v8f;

// ---------------------------------------------------------------------------
// K0: zero the output + denominators, set segment-max buffer to -inf
// ---------------------------------------------------------------------------
__global__ __launch_bounds__(256) void init_buffers(float* __restrict__ out, size_t outN,
                                                    float* __restrict__ mrow,
                                                    float* __restrict__ denom, size_t mh) {
    size_t i = (size_t)blockIdx.x * blockDim.x + threadIdx.x;
    if (i < outN) out[i] = 0.0f;
    if (i < mh) { mrow[i] = -__builtin_inff(); denom[i] = 0.0f; }
}

// ---------------------------------------------------------------------------
// K0b: convert node features f32 -> f16 (one streaming pass).
// ---------------------------------------------------------------------------
__global__ __launch_bounds__(256) void cvt_h_f16(const float* __restrict__ h,
                                                 _Float16* __restrict__ h16, size_t n) {
    size_t i = (size_t)blockIdx.x * blockDim.x + threadIdx.x;
    if (i < n) h16[i] = (_Float16)h[i];
}

// ---------------------------------------------------------------------------
// K0c: transpose+convert Wq/Wk/Wv (row-major [K][N] f32) into WT (f16, [N][K]).
// wt[m*65536 + n*256 + k] = W_m[k*256 + n].  768 KB total -> trivial.
// ---------------------------------------------------------------------------
__global__ __launch_bounds__(256) void cvt_wt_f16(const float* __restrict__ Wq,
                                                  const float* __restrict__ Wk,
                                                  const float* __restrict__ Wv,
                                                  _Float16* __restrict__ wt) {
    const int tid = blockIdx.x * 256 + threadIdx.x;      // 3*256*256 total
    const int m   = tid >> 16;
    const int rem = tid & 0xFFFF;
    const int n   = rem >> 8;
    const int k   = rem & 0xFF;
    const float* W = (m == 0) ? Wq : (m == 1) ? Wk : Wv;
    wt[tid] = (_Float16)W[k * HID + n];
}

// ---------------------------------------------------------------------------
// K1: fused Q/K/V projection via V_WMMA_F32_16X16X32_F16.
// One wave owns a 16x64 output strip (4 accumulators): A registers reused 4x.
// A (h16, row-major) and B (WT, [N][K] f16) are both loaded as aligned b128s
// matching the ISA wave32 16-bit A(16x32)/B(32x16) layouts:
//   A lane(r,half): K = k0+8*half+{0..7} and k0+16+8*half+{0..7}  (contiguous)
//   B lane(r,half): col = colBase+r, K = k0+8*half+{0..7} and k0+16+8*half+{0..7}  (same order as A)
// grid = (N/16, HID/64, 3), block = 32 (one wave; EXEC all ones, no divergence).
// ---------------------------------------------------------------------------
__global__ __launch_bounds__(32) void qkv_proj_wmma(
    const _Float16* __restrict__ h16, const _Float16* __restrict__ wt,
    const float* __restrict__ bq, const float* __restrict__ bk, const float* __restrict__ bv,
    float* __restrict__ qo, float* __restrict__ ko, float* __restrict__ vo,
    int nNodes, float qscale) {
    const int rowBase = blockIdx.x * 16;
    const int colBase = blockIdx.y * 64;

    const _Float16* __restrict__ WT = wt + (size_t)blockIdx.z * (HID * HID);
    const float* bias; float* out; float scale;
    if (blockIdx.z == 0)      { bias = bq; out = qo; scale = qscale; }
    else if (blockIdx.z == 1) { bias = bk; out = ko; scale = 1.0f; }
    else                      { bias = bv; out = vo; scale = 1.0f; }

    const int lane = threadIdx.x;
    const int r    = lane & 15;
    const int half = lane >> 4;

    const int arow_i = (rowBase + r < nNodes) ? (rowBase + r) : (nNodes - 1);
    const _Float16* __restrict__ arow = h16 + (size_t)arow_i * HID;

    v8f acc[4] = {};
    #pragma unroll
    for (int k0 = 0; k0 < HID; k0 += 32) {
        const v8h alo = *(const v8h*)(arow + k0 + 8 * half);
        const v8h ahi = *(const v8h*)(arow + k0 + 16 + 8 * half);
        const v16h a = __builtin_shufflevector(alo, ahi,
            0, 1, 2, 3, 4, 5, 6, 7, 8, 9, 10, 11, 12, 13, 14, 15);
        #pragma unroll
        for (int sub = 0; sub < 4; ++sub) {
            const _Float16* __restrict__ brow =
                WT + (size_t)(colBase + sub * 16 + r) * HID + k0 + 8 * half;
            const v8h b0 = *(const v8h*)(brow);
            const v8h b1 = *(const v8h*)(brow + 16);
            const v16h b = __builtin_shufflevector(b0, b1,
                0, 1, 2, 3, 4, 5, 6, 7, 8, 9, 10, 11, 12, 13, 14, 15);
            acc[sub] = __builtin_amdgcn_wmma_f32_16x16x32_f16(
                false, a, false, b, (short)0, acc[sub], false, false);
        }
    }

    #pragma unroll
    for (int sub = 0; sub < 4; ++sub) {
        const int   col  = colBase + sub * 16 + r;
        const float bcol = bias[col];
        #pragma unroll
        for (int vv = 0; vv < 8; ++vv) {
            const int row = rowBase + vv + 8 * half;   // C/D: VGPR v -> M = v (+8 upper half)
            if (row < nNodes)
                out[(size_t)row * HID + col] = (acc[sub][vv] + bcol) * scale;
        }
    }
}

// ---------------------------------------------------------------------------
// K2: SDDMM. One thread per (edge, head): score = <q[row], k[col]>_head * val.
// Head h owns hidden indices d*8 + h. 8 consecutive threads share one edge's
// two rows -> contiguous accesses; Q/K rows are L2-resident (154 MB < 192 MB).
// ---------------------------------------------------------------------------
__global__ __launch_bounds__(256) void sddmm_kernel(
    const float* __restrict__ q, const float* __restrict__ k,
    const float* __restrict__ val, const int* __restrict__ rows,
    const int* __restrict__ cols, float* __restrict__ score,
    float* __restrict__ mrow, int nEdges) {
    const int tid = blockIdx.x * 256 + threadIdx.x;
    if (tid >= nEdges * HEADS) return;
    const int e  = tid >> 3;
    const int hh = tid & 7;
    const int r  = rows[e];
    const int c  = cols[e];
    const float* __restrict__ qp = q + (size_t)r * HID + hh;
    const float* __restrict__ kp = k + (size_t)c * HID + hh;
    float s = 0.0f;
    #pragma unroll
    for (int d = 0; d < HD; ++d) s += qp[d * HEADS] * kp[d * HEADS];
    s *= val[e];
    score[tid] = s;
    atomicMax(&mrow[r * HEADS + hh], s);   // global_atomic_max_num_f32
}

// ---------------------------------------------------------------------------
// K3: numerator exp(score - m[row]) (overwrites score) + segment-sum denom.
// ---------------------------------------------------------------------------
__global__ __launch_bounds__(256) void softmax_num_kernel(
    const int* __restrict__ rows, float* __restrict__ score,
    const float* __restrict__ mrow, float* __restrict__ denom, int nEdges) {
    const int tid = blockIdx.x * 256 + threadIdx.x;
    if (tid >= nEdges * HEADS) return;
    const int e  = tid >> 3;
    const int hh = tid & 7;
    const int r  = rows[e];
    const float ex = __expf(score[tid] - mrow[r * HEADS + hh]);
    score[tid] = ex;
    atomicAdd(&denom[r * HEADS + hh], ex);  // global_atomic_add_f32
}

// ---------------------------------------------------------------------------
// K4: SpMM scatter: out[row, d*8+h] += attn * v[col, d*8+h].
// Sorted edge_rows => atomics to the same destination rows cluster in L2.
// ---------------------------------------------------------------------------
__global__ __launch_bounds__(256) void spmm_kernel(
    const int* __restrict__ rows, const int* __restrict__ cols,
    const float* __restrict__ score, const float* __restrict__ denom,
    const float* __restrict__ v, float* __restrict__ out, int nEdges) {
    const int tid = blockIdx.x * 256 + threadIdx.x;
    if (tid >= nEdges * HEADS) return;
    const int e  = tid >> 3;
    const int hh = tid & 7;
    const int r  = rows[e];
    const int c  = cols[e];
    const float a = score[tid] / denom[r * HEADS + hh];
    const float* __restrict__ vp = v + (size_t)c * HID + hh;
    float* __restrict__ op = out + (size_t)r * HID + hh;
    #pragma unroll
    for (int d = 0; d < HD; ++d)
        atomicAdd(&op[d * HEADS], a * vp[d * HEADS]);
}

// ---------------------------------------------------------------------------
extern "C" void kernel_launch(void* const* d_in, const int* in_sizes, int n_in,
                              void* d_out, int out_size, void* d_ws, size_t ws_size,
                              hipStream_t stream) {
    const float* h    = (const float*)d_in[0];
    const float* val  = (const float*)d_in[1];
    const float* Wq   = (const float*)d_in[2];
    const float* bq   = (const float*)d_in[3];
    const float* Wk   = (const float*)d_in[4];
    const float* bk   = (const float*)d_in[5];
    const float* Wv   = (const float*)d_in[6];
    const float* bv   = (const float*)d_in[7];
    const int*   rows = (const int*)d_in[8];
    const int*   cols = (const int*)d_in[9];

    const int N = in_sizes[0] / HID;
    const int E = in_sizes[1];
    float* out = (float*)d_out;

    // Workspace carve-up with 256B alignment. h16 (GEMM-only) aliases score
    // (written only after the GEMM has consumed h16) -> total ~183 MB.
    char*  base = (char*)d_ws;
    size_t off  = 0;
    auto alloc = [&](size_t bytes) -> char* {
        char* p = base + off;
        off += (bytes + 255) & ~(size_t)255;
        return p;
    };
    const size_t h16Bytes   = (size_t)N * HID * sizeof(_Float16);
    const size_t scoreBytes = (size_t)E * HEADS * sizeof(float);
    char* region0 = alloc(h16Bytes > scoreBytes ? h16Bytes : scoreBytes);
    _Float16* h16   = (_Float16*)region0;
    float*    score = (float*)region0;                      // alias (post-GEMM)
    _Float16* wt    = (_Float16*)alloc((size_t)3 * HID * HID * sizeof(_Float16));
    float*    q     = (float*)alloc((size_t)N * HID * sizeof(float));
    float*    k     = (float*)alloc((size_t)N * HID * sizeof(float));
    float*    v     = (float*)alloc((size_t)N * HID * sizeof(float));
    float*    mrow  = (float*)alloc((size_t)N * HEADS * sizeof(float));
    float*    denom = (float*)alloc((size_t)N * HEADS * sizeof(float));

    const size_t outN = (size_t)N * HID;
    const size_t mh   = (size_t)N * HEADS;

    init_buffers<<<(unsigned)((outN + 255) / 256), 256, 0, stream>>>(out, outN, mrow, denom, mh);
    cvt_h_f16<<<(unsigned)((outN + 255) / 256), 256, 0, stream>>>(h, h16, outN);
    cvt_wt_f16<<<(3 * HID * HID) / 256, 256, 0, stream>>>(Wq, Wk, Wv, wt);

    dim3 g1((N + 15) / 16, HID / 64, 3);
    const float qscale = 0.17677669529663687f;   // HD^-0.5 = 1/sqrt(32)
    qkv_proj_wmma<<<g1, 32, 0, stream>>>(h16, wt, bq, bk, bv, q, k, v, N, qscale);

    const int tot = E * HEADS;
    const int gs  = (tot + 255) / 256;
    sddmm_kernel<<<gs, 256, 0, stream>>>(q, k, val, rows, cols, score, mrow, E);
    softmax_num_kernel<<<gs, 256, 0, stream>>>(rows, score, mrow, denom, E);
    spmm_kernel<<<gs, 256, 0, stream>>>(rows, cols, score, denom, v, out, E);
}
